// MeshGraphNet_v2_38345468018699
// MI455X (gfx1250) — compile-verified, matrix operand order checked by reading, not run on hardware
//
#include <hip/hip_runtime.h>
#include <cstddef>
#include <cstdint>

// ---------------- constants ----------------
#define NN_NODES 50000
#define NE_EDGES 250000
#define MN_PAD   50048    // ceil(50000/128)*128
#define ME_PAD   250112   // ceil(250000/128)*128
#define HDIM     128

typedef __attribute__((ext_vector_type(16))) __bf16 v16bf;
typedef __attribute__((ext_vector_type(8)))  float  v8f;

__device__ __forceinline__ unsigned short f2bf(float f) {
  unsigned u = __builtin_bit_cast(unsigned, f);
  u += 0x7FFFu + ((u >> 16) & 1u);          // round-to-nearest-even
  return (unsigned short)(u >> 16);
}
__device__ __forceinline__ float bf2f(unsigned short s) {
  unsigned u = ((unsigned)s) << 16;
  return __builtin_bit_cast(float, u);
}

union FragU { v16bf v; uint4 q[2]; };

// ---------------- fused MLP (compile-time specialized) ----------------
struct MlpP {
  const unsigned short* A;     // bf16 input, Mpad x K0 (rows padded with zeros)
  const unsigned short* W[5];  // bf16 pre-transposed [Npad][Kpad]
  const float* bias[5];
  const float* g;
  const float* beta;
  const unsigned short* res;   // bf16 Mpad x 128 (may alias out/A)
  unsigned short* out;         // bf16 Mpad x Nlast
  float* out32;                // fp32 M x n32
  int M;                       // real rows
  int n32;
};

// One GEMM layer: 16 rows x (TILES*16) cols per wave, K-loop step 32.
template<int KL, int TILES, bool FIRST, bool RELU>
__device__ __forceinline__ void do_layer(v8f (&acc)[8],
                                         const unsigned short* A,
                                         const unsigned short* stage,
                                         const unsigned short* __restrict__ Wt,
                                         const float* __restrict__ bias,
                                         int m0, int ln, int hl) {
#pragma unroll
  for (int t = 0; t < TILES; ++t)
#pragma unroll
    for (int r = 0; r < 8; ++r) acc[t][r] = 0.f;

  for (int kk = 0; kk < KL; kk += 32) {
    const int c0 = kk + hl * 8;
    FragU a;
    if constexpr (FIRST) {
      const unsigned short* ap = A + (size_t)(m0 + ln) * KL + c0;
      a.q[0] = *(const uint4*)(ap);
      a.q[1] = *(const uint4*)(ap + 16);
    } else {
      const unsigned short* sp = stage + ln * HDIM + c0;
      a.q[0] = *(const uint4*)(sp);           // ds_load_b128
      a.q[1] = *(const uint4*)(sp + 16);
    }
#pragma unroll
    for (int t = 0; t < TILES; ++t) {
      const unsigned short* wp = Wt + (size_t)(t * 16 + ln) * KL + c0;
      FragU b;
      b.q[0] = *(const uint4*)(wp);
      b.q[1] = *(const uint4*)(wp + 16);
      acc[t] = __builtin_amdgcn_wmma_f32_16x16x32_bf16(
          false, a.v, false, b.v, (short)0, acc[t], false, false);
    }
  }

#pragma unroll
  for (int t = 0; t < TILES; ++t) {
    const float bb = bias[t * 16 + ln];
#pragma unroll
    for (int r = 0; r < 8; ++r) {
      float v = acc[t][r] + bb;
      if constexpr (RELU) v = fmaxf(v, 0.f);
      acc[t][r] = v;
    }
  }
}

// Stage wave-private 16 x (TILES*16) bf16 block into LDS and make it readable.
template<int TILES>
__device__ __forceinline__ void stage_write(const v8f (&acc)[8],
                                            unsigned short* stage, int ln, int hl) {
#pragma unroll
  for (int t = 0; t < TILES; ++t)
#pragma unroll
    for (int r = 0; r < 8; ++r)
      stage[(r + 8 * hl) * HDIM + t * 16 + ln] = f2bf(acc[t][r]);
  __builtin_amdgcn_wave_barrier();
  asm volatile("s_wait_dscnt 0" ::: "memory");   // LDS is in-order per wave
}

// LayerNorm over 128 features held in the 8 accumulator tiles.
__device__ __forceinline__ void ln_apply(v8f (&acc)[8],
                                         const float* __restrict__ g,
                                         const float* __restrict__ beta,
                                         int ln, int hl) {
#pragma unroll
  for (int r = 0; r < 8; ++r) {
    float s = 0.f, q = 0.f;
#pragma unroll
    for (int t = 0; t < 8; ++t) { float v = acc[t][r]; s += v; q += v * v; }
#pragma unroll
    for (int m = 1; m < 16; m <<= 1) {
      s += __shfl_xor(s, m, 32);
      q += __shfl_xor(q, m, 32);
    }
    const float mu  = s * (1.f / 128.f);
    const float var = q * (1.f / 128.f) - mu * mu;
    const float rs  = rsqrtf(var + 1e-5f);
#pragma unroll
    for (int t = 0; t < 8; ++t) acc[t][r] = (acc[t][r] - mu) * rs;
  }
#pragma unroll
  for (int t = 0; t < 8; ++t) {
    const float gg = g[t * 16 + ln];
    const float bb = beta[t * 16 + ln];
#pragma unroll
    for (int r = 0; r < 8; ++r) acc[t][r] = acc[t][r] * gg + bb;
  }
}

// NL<=5 layers; layer0 K=K0, others K=128; all N=128 except last (LTILES*16).
template<int NL, int K0, int LTILES, int RMASK, bool LN_, bool RES_, bool OUT_, bool O32_>
__global__ __launch_bounds__(256) void mlp_fused(MlpP p) {
  __shared__ unsigned short stage_all[8][16 * HDIM];   // 32 KB, wave-private slices
  const int lane = threadIdx.x & 31;
  const int wave = threadIdx.x >> 5;
  const int ln   = lane & 15;
  const int hl   = lane >> 4;
  unsigned short* stage = stage_all[wave];
  const int m0 = blockIdx.x * 128 + wave * 16;

  v8f acc[8];
  do_layer<K0, (NL == 1 ? LTILES : 8), true, ((RMASK >> 0) & 1) != 0>(
      acc, p.A, stage, p.W[0], p.bias[0], m0, ln, hl);
  if constexpr (NL >= 2) {
    stage_write<8>(acc, stage, ln, hl);
    do_layer<HDIM, (NL == 2 ? LTILES : 8), false, ((RMASK >> 1) & 1) != 0>(
        acc, nullptr, stage, p.W[1], p.bias[1], m0, ln, hl);
  }
  if constexpr (NL >= 3) {
    stage_write<8>(acc, stage, ln, hl);
    do_layer<HDIM, (NL == 3 ? LTILES : 8), false, ((RMASK >> 2) & 1) != 0>(
        acc, nullptr, stage, p.W[2], p.bias[2], m0, ln, hl);
  }
  if constexpr (NL >= 4) {
    stage_write<8>(acc, stage, ln, hl);
    do_layer<HDIM, (NL == 4 ? LTILES : 8), false, ((RMASK >> 3) & 1) != 0>(
        acc, nullptr, stage, p.W[3], p.bias[3], m0, ln, hl);
  }
  if constexpr (NL >= 5) {
    stage_write<8>(acc, stage, ln, hl);
    do_layer<HDIM, LTILES, false, ((RMASK >> 4) & 1) != 0>(
        acc, nullptr, stage, p.W[4], p.bias[4], m0, ln, hl);
  }

  if constexpr (LN_) ln_apply(acc, p.g, p.beta, ln, hl);

  // stage final block, then vectorized store pass with fused residual
  stage_write<LTILES>(acc, stage, ln, hl);

  constexpr int NCOLS = LTILES * 16;
  for (int r16 = 0; r16 < 16; ++r16) {
    const int row = m0 + r16;
    if constexpr (NCOLS == 128) {
      const int cc = lane * 4;                 // 4 contiguous bf16 per lane
      uint2 sv = *(const uint2*)(stage + r16 * HDIM + cc);
      float f0 = bf2f((unsigned short)(sv.x & 0xffffu));
      float f1 = bf2f((unsigned short)(sv.x >> 16));
      float f2 = bf2f((unsigned short)(sv.y & 0xffffu));
      float f3 = bf2f((unsigned short)(sv.y >> 16));
      if constexpr (RES_) {
        uint2 rv = *(const uint2*)(p.res + (size_t)row * 128 + cc);
        f0 += bf2f((unsigned short)(rv.x & 0xffffu));
        f1 += bf2f((unsigned short)(rv.x >> 16));
        f2 += bf2f((unsigned short)(rv.y & 0xffffu));
        f3 += bf2f((unsigned short)(rv.y >> 16));
      }
      if constexpr (OUT_) {
        uint2 ov;
        ov.x = (unsigned)f2bf(f0) | ((unsigned)f2bf(f1) << 16);
        ov.y = (unsigned)f2bf(f2) | ((unsigned)f2bf(f3) << 16);
        *(uint2*)(p.out + (size_t)row * 128 + cc) = ov;
      }
      if constexpr (O32_) {
        if (row < p.M) {
          if (cc + 0 < p.n32) p.out32[(size_t)row * p.n32 + cc + 0] = f0;
          if (cc + 1 < p.n32) p.out32[(size_t)row * p.n32 + cc + 1] = f1;
          if (cc + 2 < p.n32) p.out32[(size_t)row * p.n32 + cc + 2] = f2;
          if (cc + 3 < p.n32) p.out32[(size_t)row * p.n32 + cc + 3] = f3;
        }
      }
    } else {  // NCOLS == 16 (decoder tail)
      if (lane < NCOLS) {
        float v = bf2f(stage[r16 * HDIM + lane]);
        if constexpr (RES_) v += bf2f(p.res[(size_t)row * NCOLS + lane]);
        if constexpr (OUT_) p.out[(size_t)row * NCOLS + lane] = f2bf(v);
        if constexpr (O32_)
          if (row < p.M && lane < p.n32) p.out32[(size_t)row * p.n32 + lane] = v;
      }
    }
  }
}

// ---------------- utility kernels ----------------
__global__ void pack_w_kernel(const float* W, unsigned short* Wt,
                              int din, int dout, int Kpad, int Npad) {
  int i = blockIdx.x * blockDim.x + threadIdx.x;
  int tot = Npad * Kpad;
  if (i >= tot) return;
  int n = i / Kpad, k = i % Kpad;
  float v = (n < dout && k < din) ? W[(size_t)k * dout + n] : 0.f;
  Wt[i] = f2bf(v);
}

__global__ void cvt_pad_kernel(const float* src, unsigned short* dst,
                               int M, int C, int Mpad, int Cpad) {
  int i = blockIdx.x * blockDim.x + threadIdx.x;
  int tot = Mpad * Cpad;
  if (i >= tot) return;
  int r = i / Cpad, c = i % Cpad;
  unsigned short v = 0;
  if (r < M && c < C) v = f2bf(src[(size_t)r * C + c]);
  dst[i] = v;
}

__global__ void zero_f_kernel(float* p, int n) {
  int i = blockIdx.x * blockDim.x + threadIdx.x;
  if (i < n) p[i] = 0.f;
}

__global__ void pool_sum_kernel(const unsigned short* Xg, const int* batch,
                                float* gsum, float* gcnt, int M) {
  int c = threadIdx.x;  // 0..127
  int start = blockIdx.x * 256;
  int end = start + 256; if (end > M) end = M;
  float s0 = 0.f, s1 = 0.f, s2 = 0.f, s3 = 0.f;
  int c0 = 0, c1 = 0, c2 = 0, c3 = 0;
  for (int i = start; i < end; ++i) {
    int b = batch[i];
    float v = bf2f(Xg[(size_t)i * HDIM + c]);
    if (b == 0)      { s0 += v; c0++; }
    else if (b == 1) { s1 += v; c1++; }
    else if (b == 2) { s2 += v; c2++; }
    else             { s3 += v; c3++; }
  }
  atomicAdd(&gsum[0 * HDIM + c], s0);
  atomicAdd(&gsum[1 * HDIM + c], s1);
  atomicAdd(&gsum[2 * HDIM + c], s2);
  atomicAdd(&gsum[3 * HDIM + c], s3);
  if (c == 0) {
    atomicAdd(&gcnt[0], (float)c0);
    atomicAdd(&gcnt[1], (float)c1);
    atomicAdd(&gcnt[2], (float)c2);
    atomicAdd(&gcnt[3], (float)c3);
  }
}

__global__ void pool_div_kernel(const float* gsum, const float* gcnt, float* pooled) {
  int i = blockIdx.x * blockDim.x + threadIdx.x;
  if (i < 4 * HDIM) pooled[i] = gsum[i] / fmaxf(gcnt[i >> 7], 1.f);
}

__global__ void cat135_kernel(const float* node_attr, const int* batch,
                              const float* pooled, unsigned short* dst, int M) {
  int i = blockIdx.x * blockDim.x + threadIdx.x;
  int tot = MN_PAD * 160;
  if (i >= tot) return;
  int r = i / 160, c = i % 160;
  unsigned short v = 0;
  if (r < M) {
    if (c < 7)        v = f2bf(node_attr[(size_t)r * 7 + c]);
    else if (c < 135) v = f2bf(pooled[(size_t)batch[r] * HDIM + (c - 7)]);
  }
  dst[i] = v;
}

__global__ void cat256_kernel(const unsigned short* x, const float* agg,
                              const float* deg, unsigned short* dst) {
  int i = blockIdx.x * blockDim.x + threadIdx.x;
  int tot = MN_PAD * 256;
  if (i >= tot) return;
  int r = i / 256, c = i % 256;
  unsigned short v;
  if (c < HDIM) {
    v = x[(size_t)r * HDIM + c];
  } else {
    float inv = 1.f / fmaxf(deg[r], 1.f);
    v = f2bf(agg[(size_t)r * HDIM + (c - HDIM)] * inv);
  }
  dst[i] = v;
}

__global__ void deg_kernel(const int* col, float* deg, int E) {
  int i = blockIdx.x * blockDim.x + threadIdx.x;
  if (i < E) atomicAdd(&deg[col[i]], 1.f);
}

__global__ void scatter_kernel(const unsigned short* e, const int* col,
                               float* agg, int E) {
  int i = blockIdx.x * blockDim.x + threadIdx.x;
  if (i >= E * HDIM) return;
  int eid = i >> 7, c = i & 127;
  atomicAdd(&agg[(size_t)col[eid] * HDIM + c], bf2f(e[(size_t)eid * HDIM + c]));
}

// ---------------- host launcher ----------------
// d_in assumed to follow setup_inputs() dict insertion order, nested containers
// flattened recursively in insertion order (120 leaves):
//  0 node_attr  1 edge_attr  2 edge_index  3 batch
//  4..7 glob_mlp W, 8..11 b ; 12 glob_lin W, 13 b
//  14..17 node_enc W, 18..21 b, 22 g, 23 beta
//  24..27 edge_enc W, 28..31 b, 32 g, 33 beta
//  l=0..3 base=34+20l: +0..3 eW +4..7 eb +8 g +9 beta ; +10..13 nW +14..17 nb +18 g +19 beta
//  114..116 dec W, 117..119 dec b
extern "C" void kernel_launch(void* const* d_in, const int* in_sizes, int n_in,
                              void* d_out, int out_size, void* d_ws, size_t ws_size,
                              hipStream_t stream) {
  (void)in_sizes; (void)n_in; (void)out_size; (void)ws_size;
  const float* node_attr = (const float*)d_in[0];
  const float* edge_attr = (const float*)d_in[1];
  const int*   ei        = (const int*)d_in[2];
  const int*   batch     = (const int*)d_in[3];
  const int*   col       = ei + NE_EDGES;   // edge_index[1]

  size_t off = 0;
  auto alloc = [&](size_t bytes) -> char* {
    char* ptr = (char*)d_ws + off;
    off = (off + bytes + 255) & ~(size_t)255;
    return ptr;
  };

  // ---- pack all weights to bf16 transposed/padded ----
  struct WSpec { int idx, din, dout, Kpad, Npad; };
  WSpec spec[48]; int nw = 0;
  spec[nw++] = {4, 7, 128, 32, 128};
  for (int i = 0; i < 3; ++i) spec[nw++] = {5 + i, 128, 128, 128, 128};
  spec[nw++] = {12, 128, 128, 128, 128};                  // glob_lin
  spec[nw++] = {14, 135, 128, 160, 128};
  for (int i = 0; i < 3; ++i) spec[nw++] = {15 + i, 128, 128, 128, 128};
  spec[nw++] = {24, 4, 128, 32, 128};
  for (int i = 0; i < 3; ++i) spec[nw++] = {25 + i, 128, 128, 128, 128};
  for (int l = 0; l < 4; ++l) {
    int base = 34 + 20 * l;
    for (int i = 0; i < 4; ++i) spec[nw++] = {base + i, 128, 128, 128, 128};   // edge mlp
    spec[nw++] = {base + 10, 256, 128, 256, 128};                              // node mlp W0
    for (int i = 0; i < 3; ++i) spec[nw++] = {base + 11 + i, 128, 128, 128, 128};
  }
  spec[nw++] = {114, 128, 128, 128, 128};
  spec[nw++] = {115, 128, 128, 128, 128};
  spec[nw++] = {116, 128, 3, 128, 16};

  unsigned short* wt[48];
  for (int i = 0; i < nw; ++i) {
    int tot = spec[i].Kpad * spec[i].Npad;
    wt[i] = (unsigned short*)alloc((size_t)tot * 2);
    pack_w_kernel<<<(tot + 255) / 256, 256, 0, stream>>>(
        (const float*)d_in[spec[i].idx], wt[i],
        spec[i].din, spec[i].dout, spec[i].Kpad, spec[i].Npad);
  }

  // ---- workspace buffers ----
  unsigned short* na_bf  = (unsigned short*)alloc((size_t)MN_PAD * 32 * 2);
  unsigned short* ea_bf  = (unsigned short*)alloc((size_t)ME_PAD * 32 * 2);
  unsigned short* cat135 = (unsigned short*)alloc((size_t)MN_PAD * 160 * 2);
  unsigned short* xbuf   = (unsigned short*)alloc((size_t)MN_PAD * HDIM * 2);
  unsigned short* ebuf   = (unsigned short*)alloc((size_t)ME_PAD * HDIM * 2);
  unsigned short* cat256 = (unsigned short*)alloc((size_t)MN_PAD * 256 * 2);
  float* agg    = (float*)alloc((size_t)MN_PAD * HDIM * 4);
  float* deg    = (float*)alloc((size_t)MN_PAD * 4);
  float* gsum   = (float*)alloc(512 * 4);
  float* gcnt   = (float*)alloc(4 * 4);
  float* pooled = (float*)alloc(512 * 4);

  // ---- pad/convert inputs ----
  { int t = MN_PAD * 32; cvt_pad_kernel<<<(t + 255) / 256, 256, 0, stream>>>(node_attr, na_bf, NN_NODES, 7, MN_PAD, 32); }
  { int t = ME_PAD * 32; cvt_pad_kernel<<<(t + 255) / 256, 256, 0, stream>>>(edge_attr, ea_bf, NE_EDGES, 4, ME_PAD, 32); }

  const dim3 blk(256);
  const int gridN = MN_PAD / 128;
  const int gridE = ME_PAD / 128;

  // ---- GlobalEncoder: glob_mlp (4) + glob_lin fused -> xbuf ----
  {
    MlpP p{};
    p.A = na_bf;
    p.W[0]=wt[0]; p.W[1]=wt[1]; p.W[2]=wt[2]; p.W[3]=wt[3]; p.W[4]=wt[4];
    p.bias[0]=(const float*)d_in[8];  p.bias[1]=(const float*)d_in[9];
    p.bias[2]=(const float*)d_in[10]; p.bias[3]=(const float*)d_in[11];
    p.bias[4]=(const float*)d_in[13];
    p.out = xbuf; p.M = NN_NODES;
    mlp_fused<5, 32, 8, 0x7, false, false, true, false><<<gridN, blk, 0, stream>>>(p);
  }

  // ---- global mean pool + broadcast ----
  zero_f_kernel<<<2, 256, 0, stream>>>(gsum, 512);
  zero_f_kernel<<<1, 256, 0, stream>>>(gcnt, 4);
  pool_sum_kernel<<<(NN_NODES + 255) / 256, 128, 0, stream>>>(xbuf, batch, gsum, gcnt, NN_NODES);
  pool_div_kernel<<<2, 256, 0, stream>>>(gsum, gcnt, pooled);
  { int t = MN_PAD * 160; cat135_kernel<<<(t + 255) / 256, 256, 0, stream>>>(node_attr, batch, pooled, cat135, NN_NODES); }

  // ---- node encoder -> xbuf ----
  {
    MlpP p{};
    p.A = cat135;
    p.W[0]=wt[5]; p.W[1]=wt[6]; p.W[2]=wt[7]; p.W[3]=wt[8];
    p.bias[0]=(const float*)d_in[18]; p.bias[1]=(const float*)d_in[19];
    p.bias[2]=(const float*)d_in[20]; p.bias[3]=(const float*)d_in[21];
    p.g=(const float*)d_in[22]; p.beta=(const float*)d_in[23];
    p.out = xbuf; p.M = NN_NODES;
    mlp_fused<4, 160, 8, 0x7, true, false, true, false><<<gridN, blk, 0, stream>>>(p);
  }

  // ---- edge encoder -> ebuf ----
  {
    MlpP p{};
    p.A = ea_bf;
    p.W[0]=wt[9]; p.W[1]=wt[10]; p.W[2]=wt[11]; p.W[3]=wt[12];
    p.bias[0]=(const float*)d_in[28]; p.bias[1]=(const float*)d_in[29];
    p.bias[2]=(const float*)d_in[30]; p.bias[3]=(const float*)d_in[31];
    p.g=(const float*)d_in[32]; p.beta=(const float*)d_in[33];
    p.out = ebuf; p.M = NE_EDGES;
    mlp_fused<4, 32, 8, 0x7, true, false, true, false><<<gridE, blk, 0, stream>>>(p);
  }

  // ---- degree (once) ----
  zero_f_kernel<<<(MN_PAD + 255) / 256, 256, 0, stream>>>(deg, MN_PAD);
  deg_kernel<<<(NE_EDGES + 255) / 256, 256, 0, stream>>>(col, deg, NE_EDGES);

  // ---- GN layers ----
  for (int l = 0; l < 4; ++l) {
    int wb = 13 + 8 * l;   // packed-weight base
    int lb = 34 + 20 * l;  // d_in base
    // e = e + edge_mlp(e)  (in place; each row owned by exactly one wave)
    {
      MlpP p{};
      p.A = ebuf;
      p.W[0]=wt[wb]; p.W[1]=wt[wb+1]; p.W[2]=wt[wb+2]; p.W[3]=wt[wb+3];
      p.bias[0]=(const float*)d_in[lb+4]; p.bias[1]=(const float*)d_in[lb+5];
      p.bias[2]=(const float*)d_in[lb+6]; p.bias[3]=(const float*)d_in[lb+7];
      p.g=(const float*)d_in[lb+8]; p.beta=(const float*)d_in[lb+9];
      p.res = ebuf; p.out = ebuf; p.M = NE_EDGES;
      mlp_fused<4, 128, 8, 0x7, true, true, true, false><<<gridE, blk, 0, stream>>>(p);
    }
    // agg = segment_sum(e, col); divided by deg in cat256
    zero_f_kernel<<<(MN_PAD * HDIM + 255) / 256, 256, 0, stream>>>(agg, MN_PAD * HDIM);
    scatter_kernel<<<((size_t)NE_EDGES * HDIM + 255) / 256, 256, 0, stream>>>(ebuf, col, agg, NE_EDGES);
    { int t = MN_PAD * 256; cat256_kernel<<<(t + 255) / 256, 256, 0, stream>>>(xbuf, agg, deg, cat256); }
    // x = x + node_mlp([x, agg/deg])
    {
      MlpP p{};
      p.A = cat256;
      p.W[0]=wt[wb+4]; p.W[1]=wt[wb+5]; p.W[2]=wt[wb+6]; p.W[3]=wt[wb+7];
      p.bias[0]=(const float*)d_in[lb+14]; p.bias[1]=(const float*)d_in[lb+15];
      p.bias[2]=(const float*)d_in[lb+16]; p.bias[3]=(const float*)d_in[lb+17];
      p.g=(const float*)d_in[lb+18]; p.beta=(const float*)d_in[lb+19];
      p.res = xbuf; p.out = xbuf; p.M = NN_NODES;
      mlp_fused<4, 256, 8, 0x7, true, true, true, false><<<gridN, blk, 0, stream>>>(p);
    }
  }

  // ---- decoder -> d_out (fp32, 50000 x 3) ----
  {
    MlpP p{};
    p.A = xbuf;
    p.W[0]=wt[45]; p.W[1]=wt[46]; p.W[2]=wt[47];
    p.bias[0]=(const float*)d_in[117]; p.bias[1]=(const float*)d_in[118];
    p.bias[2]=(const float*)d_in[119];
    p.out32 = (float*)d_out; p.n32 = 3; p.M = NN_NODES;
    mlp_fused<3, 128, 1, 0x3, false, false, false, true><<<gridN, blk, 0, stream>>>(p);
  }
}
